// PatchedQwen2MoeSparseMoeBlockAscend_15341623181599
// MI455X (gfx1250) — matrix operand order repaired and checked by reading, not compiled, not run on hardware
//
#include <hip/hip_runtime.h>
#include <hip/hip_bf16.h>

// ---------------- problem constants ----------------
#define Hdim 2048
#define Edim 8
#define Idim 1408
#define ISdim 5632
#define Tdim 512

// tile config
#define MT 128
#define NT 64
#define KT 32
#define AS 36   // LDS stride (halfs) for A tile rows (32 data + 4 pad) == TDM pad cfg
#define BS 36   // LDS stride (halfs) for B tile cols (transposed [n][k])

typedef __attribute__((ext_vector_type(16))) __bf16        v16bf;
typedef __attribute__((ext_vector_type(8)))  float         v8f;
typedef __attribute__((ext_vector_type(8)))  unsigned int  v8u;
typedef __attribute__((ext_vector_type(4)))  unsigned int  u32x4;
typedef __attribute__((ext_vector_type(8)))  int           i32x8;
typedef __attribute__((ext_vector_type(4)))  int           i32x4;

#if defined(__HIP_DEVICE_COMPILE__) && __has_builtin(__builtin_amdgcn_tensor_load_to_lds)
#define HAVE_TDM 1
#else
#define HAVE_TDM 0
#endif

// ---------------- helpers ----------------
__device__ inline unsigned short f2bf(float f) {
    union { float f; unsigned int u; } v; v.f = f;
    unsigned int u = v.u;
    unsigned int r = u + 0x7FFFu + ((u >> 16) & 1u);   // round-to-nearest-even
    return (unsigned short)(r >> 16);
}

__device__ inline v8f wmma_bf16(v16bf a, v16bf b, v8f c) {
    return __builtin_amdgcn_wmma_f32_16x16x32_bf16(
        false, a, false, b, (short)0, c, false, false);
}

#if HAVE_TDM
// TDM: load a 128-row x 32-col bf16 tile (row stride ldA elems) from global into
// LDS with 8-byte pad after every 64-byte row (matches AS=36 halfs).
__device__ inline void tdm_load_a(unsigned int ldsOff, const void* gptr, unsigned int ldA) {
    unsigned long long ga = (unsigned long long)gptr;
    u32x4 g0;
    g0[0] = 1u;                                                  // count=1, user mode
    g0[1] = ldsOff;                                              // lds_addr (bytes)
    g0[2] = (unsigned int)ga;                                    // global_addr[31:0]
    g0[3] = (unsigned int)((ga >> 32) & 0x01FFFFFFull)           // global_addr[56:32]
            | 0x80000000u;                                       // type=2 ("image")
    i32x8 g1;
    g1[0] = (1 << 16)       // data_size = 2 bytes
          | (1 << 20)       // pad_enable
          | (3 << 22)       // pad_interval: 16 DWORDs (64B row)
          | (1 << 25);      // pad_amount: 2 DWORDs (8B pad)
    g1[1] = (int)(32u << 16);    // tensor_dim0 = 32  (bits 79:48, low half)
    g1[2] = (int)(128u << 16);   // tensor_dim0 hi=0 | tensor_dim1 = 128 (low half)
    g1[3] = (int)(32u << 16);    // tensor_dim1 hi=0 | tile_dim0 = 32
    g1[4] = 128;                 // tile_dim1 = 128 | tile_dim2 = 0
    g1[5] = (int)ldA;            // tensor_dim0_stride[31:0]
    g1[6] = 0;                   // stride hi | tensor_dim1_stride lo
    g1[7] = 0;
    i32x4 z4 = (i32x4)0;
    i32x8 z8 = (i32x8)0;
    // 6-arg form (amdgpu-toolchain / clang-23): (g0, g1, g2, g3, extra, cpol)
    __builtin_amdgcn_tensor_load_to_lds(g0, g1, z4, z4, z8, 0);
}
#endif

// cooperative fallback A-tile load (256 threads)
__device__ inline void load_a_tile_coop(unsigned short* Alds, const unsigned short* Atile,
                                        int ldA, int tid) {
#pragma unroll
    for (int i = 0; i < 8; ++i) {
        int idx2 = tid + i * 256;          // uint32 index, 2048 total
        int r = idx2 >> 4;                 // 16 uints per row
        int c = (idx2 & 15) * 2;
        *(unsigned int*)(Alds + r * AS + c) =
            *(const unsigned int*)(Atile + (long long)r * ldA + c);
    }
}

// A fragment: 16x32 bf16 per CDNA5 layout.
// lanes 0-15 (hb=0): V0..3 -> K=0..7, V4..7 -> K=16..23
// lanes 16-31 (hb=1): V0..3 -> K=8..15, V4..7 -> K=24..31
__device__ inline v16bf load_a_frag(const unsigned short* Alds, int mBase, int lane) {
    int r = lane & 15, hb = lane >> 4;
    const unsigned short* row = Alds + (mBase + r) * AS;
    v8u u;
#pragma unroll
    for (int j = 0; j < 8; ++j) {
        int k = (j < 4) ? (8 * hb + 2 * j) : (16 + 8 * hb + 2 * (j - 4));
        u[j] = *(const unsigned int*)(row + k);
    }
    return __builtin_bit_cast(v16bf, u);
}

// B fragment: 32x16 bf16 from LDS stored [n][k]; lanes 0-15: K=0..7,16..23, 16-31: K=8..15,24..31
__device__ inline v16bf load_b_frag(const unsigned short* Blds, int nBase, int lane) {
    int n = lane & 15, hb = lane >> 4;
    const unsigned short* col = Blds + (nBase + n) * BS;
    v8u u;
#pragma unroll
    for (int j = 0; j < 8; ++j) u[j] = *(const unsigned int*)(col + ((j < 4) ? (8 * hb + 2 * j) : (16 + 8 * hb + 2 * (j - 4))));
    return __builtin_bit_cast(v16bf, u);
}

// ---------------- kernel 1: router + x->bf16 + shared sigmoid gate ----------------
__global__ __launch_bounds__(256) void router_kernel(
    const float* __restrict__ X, const float* __restrict__ GateW,
    const float* __restrict__ ShexpGateW,
    unsigned short* __restrict__ Xbf, float* __restrict__ combine,
    float* __restrict__ sig, float* __restrict__ logitsOut)
{
    int t = blockIdx.x, tid = threadIdx.x, lane = tid & 31, w = tid >> 5;
    const float* x = X + (long long)t * Hdim;

    for (int h0 = tid; h0 < Hdim; h0 += 256)
        Xbf[(long long)t * Hdim + h0] = f2bf(x[h0]);

    __shared__ float slog[Edim];
    __shared__ float ssig;

    {   // wave w -> logit for expert w
        const float* gw = GateW + (long long)w * Hdim;
        float acc = 0.f;
        for (int h0 = lane; h0 < Hdim; h0 += 32) acc += x[h0] * gw[h0];
#pragma unroll
        for (int off = 16; off; off >>= 1) acc += __shfl_xor(acc, off, 32);
        if (lane == 0) slog[w] = acc;
    }
    if (w == 0) {   // shared-expert sigmoid gate
        float a2 = 0.f;
        for (int h0 = lane; h0 < Hdim; h0 += 32) a2 += x[h0] * ShexpGateW[h0];
#pragma unroll
        for (int off = 16; off; off >>= 1) a2 += __shfl_xor(a2, off, 32);
        if (lane == 0) ssig = 1.f / (1.f + __expf(-a2));
    }
    __syncthreads();

    if (tid == 0) {
        sig[t] = ssig;
        float lg[Edim], p[Edim];
        float mx = -1e30f;
#pragma unroll
        for (int e = 0; e < Edim; ++e) {
            lg[e] = slog[e];
            logitsOut[t * Edim + e] = lg[e];
            mx = fmaxf(mx, lg[e]);
        }
        float se = 0.f;
#pragma unroll
        for (int e = 0; e < Edim; ++e) { p[e] = __expf(lg[e] - mx); se += p[e]; }
        float inv = 1.f / se;
#pragma unroll
        for (int e = 0; e < Edim; ++e) p[e] *= inv;
        int i0 = 0;
#pragma unroll
        for (int e = 1; e < Edim; ++e) if (p[e] > p[i0]) i0 = e;
        int i1 = (i0 == 0) ? 1 : 0;
#pragma unroll
        for (int e = 0; e < Edim; ++e)
            if (e != i0 && p[e] > p[i1]) i1 = e;
        float s2 = p[i0] + p[i1];
#pragma unroll
        for (int e = 0; e < Edim; ++e) combine[t * Edim + e] = 0.f;
        combine[t * Edim + i0] = p[i0] / s2;
        combine[t * Edim + i1] = p[i1] / s2;
    }
}

// ---------------- kernel 2: gate/up GEMM + SwiGLU (+combine fold) ----------------
template <int N, bool HASCOMB>
__global__ __launch_bounds__(256) void gemm_silu_kernel(
    const unsigned short* __restrict__ Xbf,
    const float* __restrict__ BgBase, const float* __restrict__ BuBase,
    unsigned short* __restrict__ HoutBase,
    const float* __restrict__ combine)
{
    __shared__ unsigned short Alds[MT * AS];
    __shared__ unsigned short Bg_lds[NT * BS];
    __shared__ unsigned short Bu_lds[NT * BS];

    int e = blockIdx.z;
    const float* Bg = BgBase + (long long)e * ((long long)Hdim * N);
    const float* Bu = BuBase + (long long)e * ((long long)Hdim * N);
    unsigned short* Hout = HoutBase + (long long)e * ((long long)Tdim * N);

    int tid = threadIdx.x, lane = tid & 31, w = tid >> 5;
    int waveM = w & 3, waveN = w >> 2;           // 4 x 2 waves
    int tBase = blockIdx.y * MT;
    int nBase = blockIdx.x * NT;

    const unsigned short* agp = Xbf + (long long)tBase * Hdim;
#if HAVE_TDM
    unsigned int aOff = (unsigned int)(unsigned long long)&Alds[0];
#endif
    int bk = tid >> 6, bn = tid & 63;
    const float* bgp = Bg + (long long)bk * N + nBase + bn;
    const float* bup = Bu + (long long)bk * N + nBase + bn;
    int bL = bn * BS + bk;

    v8f accg[2][2] = {};
    v8f accu[2][2] = {};

    for (int k0 = 0; k0 < Hdim; k0 += KT) {
#if HAVE_TDM
        if (w == 0) tdm_load_a(aOff, agp, Hdim);   // async DMA of A tile
#else
        load_a_tile_coop(Alds, agp, Hdim, tid);
#endif
#pragma unroll
        for (int i = 0; i < 8; ++i) Bg_lds[bL + 4 * i] = f2bf(bgp[(long long)i * (4 * N)]);
#pragma unroll
        for (int i = 0; i < 8; ++i) Bu_lds[bL + 4 * i] = f2bf(bup[(long long)i * (4 * N)]);
        if (lane == 0 && k0 + KT < Hdim) {
            __builtin_prefetch(bgp + (long long)KT * N, 0, 1);
            __builtin_prefetch(bup + (long long)KT * N, 0, 1);
        }
#if HAVE_TDM
        if (w == 0) __builtin_amdgcn_s_wait_tensorcnt(0);
#endif
        __syncthreads();

        v16bf a0 = load_a_frag(Alds, waveM * 32, lane);
        v16bf a1 = load_a_frag(Alds, waveM * 32 + 16, lane);
        v16bf bg0 = load_b_frag(Bg_lds, waveN * 32, lane);
        v16bf bg1 = load_b_frag(Bg_lds, waveN * 32 + 16, lane);
        v16bf bu0 = load_b_frag(Bu_lds, waveN * 32, lane);
        v16bf bu1 = load_b_frag(Bu_lds, waveN * 32 + 16, lane);

        accg[0][0] = wmma_bf16(a0, bg0, accg[0][0]);
        accg[0][1] = wmma_bf16(a0, bg1, accg[0][1]);
        accg[1][0] = wmma_bf16(a1, bg0, accg[1][0]);
        accg[1][1] = wmma_bf16(a1, bg1, accg[1][1]);
        accu[0][0] = wmma_bf16(a0, bu0, accu[0][0]);
        accu[0][1] = wmma_bf16(a0, bu1, accu[0][1]);
        accu[1][0] = wmma_bf16(a1, bu0, accu[1][0]);
        accu[1][1] = wmma_bf16(a1, bu1, accu[1][1]);
        __syncthreads();

        agp += KT;
        bgp += (long long)KT * N;
        bup += (long long)KT * N;
    }

    int nn = lane & 15, hb = lane >> 4;
#pragma unroll
    for (int sm = 0; sm < 2; ++sm) {
#pragma unroll
        for (int r = 0; r < 8; ++r) {
            int mg = tBase + waveM * 32 + sm * 16 + r + 8 * hb;
            float scale = HASCOMB ? combine[mg * Edim + e] : 1.0f;
#pragma unroll
            for (int sn = 0; sn < 2; ++sn) {
                int ng = nBase + waveN * 32 + sn * 16 + nn;
                float g = accg[sm][sn][r];
                float u = accu[sm][sn][r];
                float hv = g * (1.f / (1.f + __expf(-g))) * u * scale;
                Hout[(long long)mg * N + ng] = f2bf(hv);
            }
        }
    }
}

// ---------------- kernel 3: down-proj over all experts + shared, fused output ----
__global__ __launch_bounds__(256) void down_kernel(
    const unsigned short* __restrict__ Hexp,   // [E][T][I] bf16 (combine folded)
    const float* __restrict__ Wdown,           // [E][I][H] f32
    const unsigned short* __restrict__ Hsh,    // [T][IS] bf16
    const float* __restrict__ Wshdown,         // [IS][H] f32
    const float* __restrict__ sig,             // [T]
    float* __restrict__ Out)                   // [T][H]
{
    __shared__ unsigned short Alds[MT * AS];
    __shared__ unsigned short Blds[NT * BS];

    int tid = threadIdx.x, lane = tid & 31, w = tid >> 5;
    int waveM = w & 3, waveN = w >> 2;
    int tBase = blockIdx.y * MT;
    int nBase = blockIdx.x * NT;

#if HAVE_TDM
    unsigned int aOff = (unsigned int)(unsigned long long)&Alds[0];
#endif
    int bk = tid >> 6, bn = tid & 63;
    int bL = bn * BS + bk;

    v8f acc[2][2] = {};    // expert sum
    v8f accs[2][2] = {};   // shared expert

    for (int e = 0; e < Edim; ++e) {
        const unsigned short* agp = Hexp + ((long long)e * Tdim + tBase) * Idim;
        const float* bp = Wdown + (long long)e * Idim * Hdim
                        + (long long)bk * Hdim + nBase + bn;
        for (int k0 = 0; k0 < Idim; k0 += KT) {
#if HAVE_TDM
            if (w == 0) tdm_load_a(aOff, agp, Idim);
#else
            load_a_tile_coop(Alds, agp, Idim, tid);
#endif
#pragma unroll
            for (int i = 0; i < 8; ++i) Blds[bL + 4 * i] = f2bf(bp[i * (4 * Hdim)]);
            if (lane == 0 && k0 + KT < Idim)
                __builtin_prefetch(bp + KT * Hdim, 0, 1);
#if HAVE_TDM
            if (w == 0) __builtin_amdgcn_s_wait_tensorcnt(0);
#endif
            __syncthreads();
            v16bf a0 = load_a_frag(Alds, waveM * 32, lane);
            v16bf a1 = load_a_frag(Alds, waveM * 32 + 16, lane);
            v16bf b0 = load_b_frag(Blds, waveN * 32, lane);
            v16bf b1 = load_b_frag(Blds, waveN * 32 + 16, lane);
            acc[0][0] = wmma_bf16(a0, b0, acc[0][0]);
            acc[0][1] = wmma_bf16(a0, b1, acc[0][1]);
            acc[1][0] = wmma_bf16(a1, b0, acc[1][0]);
            acc[1][1] = wmma_bf16(a1, b1, acc[1][1]);
            __syncthreads();
            agp += KT;
            bp += KT * Hdim;
        }
    }
    {   // shared expert down-proj
        const unsigned short* agp = Hsh + (long long)tBase * ISdim;
        const float* bp = Wshdown + (long long)bk * Hdim + nBase + bn;
        for (int k0 = 0; k0 < ISdim; k0 += KT) {
#if HAVE_TDM
            if (w == 0) tdm_load_a(aOff, agp, ISdim);
#else
            load_a_tile_coop(Alds, agp, ISdim, tid);
#endif
#pragma unroll
            for (int i = 0; i < 8; ++i) Blds[bL + 4 * i] = f2bf(bp[i * (4 * Hdim)]);
            if (lane == 0 && k0 + KT < ISdim)
                __builtin_prefetch(bp + KT * Hdim, 0, 1);
#if HAVE_TDM
            if (w == 0) __builtin_amdgcn_s_wait_tensorcnt(0);
#endif
            __syncthreads();
            v16bf a0 = load_a_frag(Alds, waveM * 32, lane);
            v16bf a1 = load_a_frag(Alds, waveM * 32 + 16, lane);
            v16bf b0 = load_b_frag(Blds, waveN * 32, lane);
            v16bf b1 = load_b_frag(Blds, waveN * 32 + 16, lane);
            accs[0][0] = wmma_bf16(a0, b0, accs[0][0]);
            accs[0][1] = wmma_bf16(a0, b1, accs[0][1]);
            accs[1][0] = wmma_bf16(a1, b0, accs[1][0]);
            accs[1][1] = wmma_bf16(a1, b1, accs[1][1]);
            __syncthreads();
            agp += KT;
            bp += KT * Hdim;
        }
    }

    int nn = lane & 15, hb = lane >> 4;
#pragma unroll
    for (int sm = 0; sm < 2; ++sm) {
#pragma unroll
        for (int r = 0; r < 8; ++r) {
            int mg = tBase + waveM * 32 + sm * 16 + r + 8 * hb;
            float sv = sig[mg];
#pragma unroll
            for (int sn = 0; sn < 2; ++sn) {
                int ng = nBase + waveN * 32 + sn * 16 + nn;
                Out[(long long)mg * Hdim + ng] = acc[sm][sn][r] + sv * accs[sm][sn][r];
            }
        }
    }
}

// ---------------- launcher ----------------
extern "C" void kernel_launch(void* const* d_in, const int* in_sizes, int n_in,
                              void* d_out, int out_size, void* d_ws, size_t ws_size,
                              hipStream_t stream) {
    (void)in_sizes; (void)n_in; (void)out_size; (void)ws_size;

    const float* x          = (const float*)d_in[0];
    const float* gate_w     = (const float*)d_in[1];
    const float* w_gate     = (const float*)d_in[2];
    const float* w_up       = (const float*)d_in[3];
    const float* w_down     = (const float*)d_in[4];
    const float* sh_gate    = (const float*)d_in[5];
    const float* sh_up      = (const float*)d_in[6];
    const float* sh_down    = (const float*)d_in[7];
    const float* shexp_gate = (const float*)d_in[8];
    float* out = (float*)d_out;

    char* ws = (char*)d_ws;
    size_t off = 0;
    unsigned short* xbf  = (unsigned short*)(ws + off); off += (size_t)Tdim * Hdim * 2;
    unsigned short* hexp = (unsigned short*)(ws + off); off += (size_t)Edim * Tdim * Idim * 2;
    unsigned short* hsh  = (unsigned short*)(ws + off); off += (size_t)Tdim * ISdim * 2;
    float* combine       = (float*)(ws + off);          off += (size_t)Tdim * Edim * 4;
    float* sig           = (float*)(ws + off);          off += (size_t)Tdim * 4;

    router_kernel<<<Tdim, 256, 0, stream>>>(x, gate_w, shexp_gate, xbf, combine, sig,
                                            out + (size_t)Tdim * Hdim);

    gemm_silu_kernel<Idim, true><<<dim3(Idim / NT, Tdim / MT, Edim), 256, 0, stream>>>(
        xbf, w_gate, w_up, hexp, combine);

    gemm_silu_kernel<ISdim, false><<<dim3(ISdim / NT, Tdim / MT, 1), 256, 0, stream>>>(
        xbf, sh_gate, sh_up, hsh, nullptr);

    down_kernel<<<dim3(Hdim / NT, Tdim / MT), 256, 0, stream>>>(
        hexp, w_down, hsh, sh_down, sig, out);
}
